// SimpleGCN_32968168964259
// MI455X (gfx1250) — matrix operand order repaired and checked by reading, not compiled, not run on hardware
//
#include <hip/hip_runtime.h>
#include <hip/hip_bf16.h>

typedef __attribute__((ext_vector_type(16))) __bf16 v16bf;
typedef __attribute__((ext_vector_type(8)))  __bf16 v8bf;
typedef __attribute__((ext_vector_type(4)))  __bf16 v4bf;
typedef __attribute__((ext_vector_type(8)))  float  v8f;

#define IN_C 256
#define HID  256
#define OUTC 10
#define NGRAPH 64
#define KSTEPS (IN_C / 32)   // 8
#define NTILES (HID / 16)    // 16

// ---------------- utility kernels ----------------

__global__ void zero_f32(float* __restrict__ p, int n) {
    int i = blockIdx.x * blockDim.x + threadIdx.x;
    if (i < n) p[i] = 0.0f;
}

__global__ void fill_f32(float* __restrict__ p, float v, int n) {
    int i = blockIdx.x * blockDim.x + threadIdx.x;
    if (i < n) p[i] = v;
}

__global__ void cvt_f32_bf16(const float* __restrict__ src, __bf16* __restrict__ dst, int n) {
    int i = blockIdx.x * blockDim.x + threadIdx.x;
    if (i < n) dst[i] = (__bf16)src[i];
}

// Pack W1 (f32, [256][256] row-major) into WMMA B-fragment layout (bf16):
//   w1p[(((ntile*KSTEPS + ks)*32 + lane)*16 + i]
//     = W1[(ks*32 + (lane>>4)*16 + i) * HID + ntile*16 + (lane&15)]
// so each (ntile, ks, lane) owns a contiguous 32B fragment slice.
__global__ void pack_W1_frag(const float* __restrict__ W1, __bf16* __restrict__ w1p) {
    int idx = blockIdx.x * blockDim.x + threadIdx.x;   // 16*8*32*16 = 65536
    if (idx >= NTILES * KSTEPS * 32 * 16) return;
    int i     = idx & 15;
    int lane  = (idx >> 4) & 31;
    int ks    = (idx >> 9) & (KSTEPS - 1);
    int ntile = idx >> 12;
    int hs = lane >> 4, lr = lane & 15;
    int krow = ks * 32 + hs * 16 + i;
    int col  = ntile * 16 + lr;
    w1p[idx] = (__bf16)W1[(size_t)krow * HID + col];
}

// degree accumulation over destination nodes (self-loop handled by init=1.0)
__global__ void deg_accum(const int* __restrict__ ei, float* __restrict__ deg, int E) {
    int e = blockIdx.x * blockDim.x + threadIdx.x;
    if (e < E) atomicAdd(&deg[ei[E + e]], 1.0f);
}

__global__ void rsqrt_inplace(float* __restrict__ p, int n) {
    int i = blockIdx.x * blockDim.x + threadIdx.x;
    if (i < n) p[i] = rsqrtf(p[i]);
}

// ---------------- WMMA GEMM: h[N,256] = bf16( x[N,256] @ W1[256,256] ) ----------------
// Block = 256 threads = 8 waves. Block b owns M-tile rows [16b, 16b+16).
// Wave w owns N-tiles {2w, 2w+1}. K-loop: 8 unrolled steps of 32 (bf16).
__global__ void __launch_bounds__(256)
gemm_xW1_wmma(const __bf16* __restrict__ xb, const __bf16* __restrict__ w1p,
              __bf16* __restrict__ hb, int N) {
    const int wave = threadIdx.x >> 5;
    const int lane = threadIdx.x & 31;
    const int hs   = lane >> 4;     // half-wave select
    const int lr   = lane & 15;     // lane-in-half
    const int mBase = blockIdx.x * 16;
    const int t0 = wave * 2;        // first owned N-tile
    const int n0 = t0 * 16;
    const int n1 = n0 + 16;

    v8f acc0 = {};
    v8f acc1 = {};

    int arow = mBase + lr;
    if (arow >= N) arow = N - 1;          // clamp loads only (stores are guarded)
    const __bf16* arowp = xb + (size_t)arow * IN_C + hs * 16;
    const __bf16* bp0 = w1p + ((size_t)(t0    ) * KSTEPS * 32 + lane) * 16;
    const __bf16* bp1 = w1p + ((size_t)(t0 + 1) * KSTEPS * 32 + lane) * 16;

    #pragma unroll
    for (int ks = 0; ks < KSTEPS; ++ks) {
        // A fragment: 16x32 bf16, contiguous K (half-wave selects K half, k = 16*hs + i)
        const __bf16* ap = arowp + ks * 32;
        v8bf alo = *reinterpret_cast<const v8bf*>(ap);
        v8bf ahi = *reinterpret_cast<const v8bf*>(ap + 8);
        v16bf a;
        #pragma unroll
        for (int i = 0; i < 8; ++i) { a[i] = alo[i]; a[8 + i] = ahi[i]; }

        // B fragments: contiguous 32B per lane from pre-packed W1
        v16bf b0 = *reinterpret_cast<const v16bf*>(bp0 + (size_t)ks * 32 * 16);
        v16bf b1 = *reinterpret_cast<const v16bf*>(bp1 + (size_t)ks * 32 * 16);

        acc0 = __builtin_amdgcn_wmma_f32_16x16x32_bf16(
            false, a, false, b0, (short)0, acc0, false, false);
        acc1 = __builtin_amdgcn_wmma_f32_16x16x32_bf16(
            false, a, false, b1, (short)0, acc1, false, false);
    }

    // D layout: VGPR r -> row (hs*8 + r), col = lr. Uniform branch on full tile.
    __bf16* hrow = hb + (size_t)(mBase + hs * 8) * HID + lr;
    if (mBase + 16 <= N) {
        #pragma unroll
        for (int r = 0; r < 8; ++r) {
            hrow[(size_t)r * HID + n0] = (__bf16)acc0[r];
            hrow[(size_t)r * HID + n1] = (__bf16)acc1[r];
        }
    } else {
        #pragma unroll
        for (int r = 0; r < 8; ++r) {
            if (mBase + hs * 8 + r < N) {
                hrow[(size_t)r * HID + n0] = (__bf16)acc0[r];
                hrow[(size_t)r * HID + n1] = (__bf16)acc1[r];
            }
        }
    }
}

// ---------------- edge scatter: agg[dst] += h[src] * dinv[src]*dinv[dst] ----------------
// One block (64 threads) per edge; thread handles 4 channels. e >= E are self-loops.
__global__ void __launch_bounds__(64)
scatter_edges(const int* __restrict__ ei, const __bf16* __restrict__ hb,
              const float* __restrict__ dinv, float* __restrict__ agg, int E) {
    int e = blockIdx.x;
    int srcn, dstn;
    if (e < E) { srcn = ei[e]; dstn = ei[E + e]; }
    else       { srcn = dstn = e - E; }
    const float nrm = dinv[srcn] * dinv[dstn];
    const int c = threadIdx.x * 4;
    const v4bf hv = *reinterpret_cast<const v4bf*>(hb + (size_t)srcn * HID + c);
    float* ap = agg + (size_t)dstn * HID + c;
    #pragma unroll
    for (int i = 0; i < 4; ++i)
        atomicAdd(ap + i, (float)hv[i] * nrm);
}

// ---------------- relu(agg + b1) then mean-pool (accumulate sums + counts) ----------------
__global__ void __launch_bounds__(64)
relu_pool(const float* __restrict__ agg, const float* __restrict__ b1,
          const int* __restrict__ batch, float* __restrict__ pooled,
          float* __restrict__ counts, int N) {
    int n = blockIdx.x;
    if (n >= N) return;
    int g = batch[n];
    int c = threadIdx.x * 4;
    #pragma unroll
    for (int i = 0; i < 4; ++i) {
        float v = agg[(size_t)n * HID + c + i] + b1[c + i];
        v = fmaxf(v, 0.0f);
        atomicAdd(&pooled[(size_t)g * HID + c + i], v);
    }
    if (threadIdx.x == 0) atomicAdd(&counts[g], 1.0f);
}

// ---------------- final: out[g,o] = (pooled[g]/count[g]) @ W2 + b2 ----------------
__global__ void final_linear(const float* __restrict__ pooled, const float* __restrict__ counts,
                             const float* __restrict__ W2, const float* __restrict__ b2,
                             float* __restrict__ out) {
    int idx = blockIdx.x * blockDim.x + threadIdx.x;
    if (idx >= NGRAPH * OUTC) return;
    int g = idx / OUTC, o = idx % OUTC;
    float s = 0.0f;
    for (int c = 0; c < HID; ++c)
        s += pooled[(size_t)g * HID + c] * W2[c * OUTC + o];
    out[idx] = s / fmaxf(counts[g], 1.0f) + b2[o];
}

// ---------------- launch ----------------
extern "C" void kernel_launch(void* const* d_in, const int* in_sizes, int n_in,
                              void* d_out, int out_size, void* d_ws, size_t ws_size,
                              hipStream_t stream) {
    const float* x    = (const float*)d_in[0];
    const int*   ei   = (const int*)d_in[1];   // [2,E] flat
    const int*   batch= (const int*)d_in[2];
    const float* W1   = (const float*)d_in[3];
    const float* b1   = (const float*)d_in[4];
    const float* W2   = (const float*)d_in[5];
    const float* b2   = (const float*)d_in[6];
    float* out = (float*)d_out;

    const int N = in_sizes[0] / IN_C;
    const int E = in_sizes[1] / 2;

    // workspace layout
    char* w = (char*)d_ws;
    size_t off = 0;
    auto alloc = [&](size_t bytes) { void* p = w + off; off = (off + bytes + 255) & ~(size_t)255; return p; };
    __bf16* xb   = (__bf16*)alloc((size_t)N * IN_C * sizeof(__bf16));
    __bf16* w1p  = (__bf16*)alloc((size_t)IN_C * HID * sizeof(__bf16));  // packed B fragments
    __bf16* hb   = (__bf16*)alloc((size_t)N * HID * sizeof(__bf16));
    float*  deg  = (float*)alloc((size_t)N * sizeof(float));             // becomes dinv
    float*  agg  = (float*)alloc((size_t)N * HID * sizeof(float));
    float*  pooled = (float*)alloc((size_t)NGRAPH * HID * sizeof(float));
    float*  counts = (float*)alloc((size_t)NGRAPH * sizeof(float));
    (void)ws_size;

    const int B = 256;
    const int nx = N * IN_C;
    const int nw = IN_C * HID;

    // 1) precision conversion + B-fragment packing
    cvt_f32_bf16<<<(nx + B - 1) / B, B, 0, stream>>>(x, xb, nx);
    pack_W1_frag<<<(nw + B - 1) / B, B, 0, stream>>>(W1, w1p);

    // 2) symmetric-norm degree: init 1.0 (self-loop), accumulate, rsqrt
    fill_f32<<<(N + B - 1) / B, B, 0, stream>>>(deg, 1.0f, N);
    deg_accum<<<(E + B - 1) / B, B, 0, stream>>>(ei, deg, E);
    rsqrt_inplace<<<(N + B - 1) / B, B, 0, stream>>>(deg, N);

    // 3) h = x @ W1   (bf16 WMMA, f32 accumulate, bf16 store)
    gemm_xW1_wmma<<<(N + 15) / 16, 256, 0, stream>>>(xb, w1p, hb, N);

    // 4) agg = scatter-add of normalized messages (incl. self-loops)
    const int nagg = N * HID;
    zero_f32<<<(nagg + B - 1) / B, B, 0, stream>>>(agg, nagg);
    scatter_edges<<<E + N, 64, 0, stream>>>(ei, hb, deg, agg, E);

    // 5) relu(agg + b1), mean pool per graph
    const int npz = NGRAPH * HID + NGRAPH;     // pooled and counts are adjacent
    zero_f32<<<(npz + B - 1) / B, B, 0, stream>>>(pooled, npz);
    relu_pool<<<N, 64, 0, stream>>>(agg, b1, batch, pooled, counts, N);

    // 6) final linear
    final_linear<<<(NGRAPH * OUTC + B - 1) / B, B, 0, stream>>>(pooled, counts, W2, b2, out);
}
